// HybridClassifier_70360154243357
// MI455X (gfx1250) — matrix operand order repaired and checked by reading, not compiled, not run on hardware
//
#include <hip/hip_runtime.h>
#include <hip/hip_bf16.h>
#include <cstdint>

// ---------------- CDNA5 WMMA types ----------------
typedef __attribute__((ext_vector_type(16))) _Float16 v16h;
typedef __attribute__((ext_vector_type(8)))  float    v8f;

struct __align__(16) U4 { uint32_t a, b, c, d; };
union Frag16 { v16h v; _Float16 h[16]; U4 q[2]; };
union Acc8   { v8f  v; float    f[8];          };

#define NBATCH 256

// ---------------- converters ----------------
// Wfc [50][65536] fp32 (feature index = ic*1024 + s, CHW-flat) ->
// f16 permuted to NHWC-flat feature order (s*64 + ic) to match XA layout.
__global__ void k_wfc_perm(const float* __restrict__ src, _Float16* __restrict__ dst) {
    int i = blockIdx.x * blockDim.x + threadIdx.x;        // over 50*65536
    int k  = i & 65535, oc = i >> 16;
    int ic = k >> 10, s = k & 1023;
    dst[((size_t)oc << 16) + (size_t)s * 64 + ic] = (_Float16)src[i];
}

// W2: [64 oc][32 ic][3][3] fp32  ->  W2t: [9 tap][64 oc][32 ic] f16 (ic contiguous for B frags)
__global__ void k_w2_transpose(const float* __restrict__ W2, _Float16* __restrict__ W2t) {
    int i = blockIdx.x * blockDim.x + threadIdx.x;        // 0..18431
    if (i >= 64 * 32 * 9) return;
    int t = i % 9, rem = i / 9, ic = rem % 32, oc = rem / 32;
    W2t[t * 2048 + oc * 32 + ic] = (_Float16)W2[i];
}

// ---------------- conv1 (3->32) + relu + maxpool2 -> P1 f16 NHWC [256,64,64,32] ------------
__global__ void k_conv1_pool(const float* __restrict__ img, const float* __restrict__ W1,
                             const float* __restrict__ b1, _Float16* __restrict__ P1) {
    __shared__ float sW[32 * 27];
    __shared__ float sB[32];
    for (int i = threadIdx.x; i < 32 * 27; i += blockDim.x) sW[i] = W1[i];
    if (threadIdx.x < 32) sB[threadIdx.x] = b1[threadIdx.x];
    __syncthreads();
    int idx = blockIdx.x * blockDim.x + threadIdx.x;       // [b, ph, pw, oc] -> NHWC store
    int oc = idx & 31, t1 = idx >> 5;
    int pw = t1 & 63, t2 = t1 >> 6;
    int ph = t2 & 63, b  = t2 >> 6;
    const float* wk = &sW[oc * 27];
    float m = -1e30f;
    for (int dy = 0; dy < 2; ++dy)
    for (int dx = 0; dx < 2; ++dx) {
        int oh = ph * 2 + dy, ow = pw * 2 + dx;
        float acc = 0.f;
        for (int c = 0; c < 3; ++c)
        for (int ky = 0; ky < 3; ++ky) {
            int iy = oh + ky - 1;
            if ((unsigned)iy >= 128u) continue;
            const float* row = &img[((size_t)(b * 3 + c) * 128 + iy) * 128];
            for (int kx = 0; kx < 3; ++kx) {
                int ix = ow + kx - 1;
                if ((unsigned)ix < 128u) acc += row[ix] * wk[(c * 3 + ky) * 3 + kx];
            }
        }
        m = fmaxf(m, acc);
    }
    P1[idx] = (_Float16)fmaxf(m + sB[oc], 0.f);            // relu(max+b) == max(relu)
}

// ---------------- conv2 (32->64) via WMMA implicit GEMM, fused relu+maxpool2 ----------------
// tile: 2 rows x 8 cols of the 64x64 conv output (M=16), N=64 ocs via 4 waves, K=32 per tap.
// NHWC P1: A fragment = two global_load_b128 per tap.
__global__ void __launch_bounds__(128) k_conv2_wmma(const _Float16* __restrict__ P1,
                                                    const _Float16* __restrict__ W2t,
                                                    const float* __restrict__ b2,
                                                    _Float16* __restrict__ P2) {
    __shared__ __align__(16) _Float16 sW[9 * 64 * 32];     // 36 KB, whole weight set
    for (int i = threadIdx.x; i < (9 * 64 * 32) / 8; i += blockDim.x)
        ((U4*)sW)[i] = ((const U4*)W2t)[i];
    __syncthreads();

    int wave = threadIdx.x >> 5, lane = threadIdx.x & 31;
    int hi = (lane >= 16) ? 1 : 0;
    int lm = lane & 15;
    int blk = blockIdx.x;
    int mt  = blk & 255, b = blk >> 8;
    int oh0 = (mt >> 3) * 2;                               // even row
    int ow0 = (mt & 7) * 8;                                // col block of 8
    int oc  = wave * 16 + lm;

    Acc8 acc;
    #pragma unroll
    for (int r = 0; r < 8; ++r) acc.f[r] = 0.f;

    for (int t = 0; t < 9; ++t) {
        int ky = t / 3, kx = t % 3;
        int ih = oh0 + (lm >> 3) + ky - 1;                 // M = row*8+col, row=lm>>3
        int iw = ow0 + (lm & 7) + kx - 1;
        Frag16 a;
        if ((unsigned)ih < 64u && (unsigned)iw < 64u) {
            // NHWC: 32 contiguous ic halfs at this pixel; A layout wants
            // halfs {hi*8 .. +7} (e=0..7 -> ic=e+hi*8) and {16+hi*8 .. +7} (e=8..15).
            const U4* ap = (const U4*)&P1[((((size_t)b * 64 + ih) * 64) + iw) * 32 + hi * 8];
            a.q[0] = ap[0];
            a.q[1] = ap[2];
        } else {
            #pragma unroll
            for (int e = 0; e < 16; ++e) a.h[e] = (_Float16)0.f;
        }
        Frag16 bf;                                          // B: K=hi*8+{0..7}, 16+hi*8+{0..7}
        const U4* wp = (const U4*)&sW[t * 2048 + oc * 32 + hi * 8];
        bf.q[0] = wp[0]; bf.q[1] = wp[2];
        acc.v = __builtin_amdgcn_wmma_f32_16x16x32_f16(false, a.v, false, bf.v,
                                                       (short)0, acc.v, false, false);
    }

    // fused 2x2 maxpool: cols pair within lane, rows pair across lane halves
    float bias = b2[oc];
    float pm[4];
    #pragma unroll
    for (int j = 0; j < 4; ++j) {
        float v = fmaxf(acc.f[2 * j], acc.f[2 * j + 1]);
        float o = __shfl_xor(v, 16, 32);
        pm[j] = fmaxf(v, o);
    }
    if (!hi) {
        int ph = oh0 >> 1, pw0 = ow0 >> 1;
        #pragma unroll
        for (int j = 0; j < 4; ++j)                        // NHWC P2 [b,ph,pw,oc]
            P2[((((size_t)b * 32 + ph) * 32) + pw0 + j) * 64 + oc] =
                (_Float16)fmaxf(pm[j] + bias, 0.f);
    }
}

// ---------------- 7x7 attention conv (64->1) + sigmoid + scale -> XA (NHWC) ----------------
__global__ void k_attn(const _Float16* __restrict__ P2, const float* __restrict__ Wa,
                       const float* __restrict__ ba, _Float16* __restrict__ XA) {
    __shared__ float sWt[49 * 64];                         // transposed: [tap][ic]
    for (int i = threadIdx.x; i < 64 * 49; i += blockDim.x) {
        int ic = i / 49, t = i % 49;
        sWt[t * 64 + ic] = Wa[i];
    }
    __syncthreads();
    int idx = blockIdx.x * blockDim.x + threadIdx.x;       // [b, oh, ow]
    int ow = idx & 31, t1 = idx >> 5, oh = t1 & 31, b = t1 >> 5;
    float acc = ba[0];
    for (int ky = 0; ky < 7; ++ky) {
        int iy = oh + ky - 3;
        if ((unsigned)iy >= 32u) continue;
        for (int kx = 0; kx < 7; ++kx) {
            int ix = ow + kx - 3;
            if ((unsigned)ix >= 32u) continue;
            const _Float16* px = &P2[((((size_t)b * 32 + iy) * 32) + ix) * 64];
            const float* wk = &sWt[(ky * 7 + kx) * 64];
            #pragma unroll 8
            for (int ic = 0; ic < 64; ++ic) acc += (float)px[ic] * wk[ic];
        }
    }
    float attn = 1.f / (1.f + expf(-acc));
    const _Float16* src = &P2[((((size_t)b * 32 + oh) * 32) + ow) * 64];
    _Float16*       dst = &XA[((((size_t)b * 32 + oh) * 32) + ow) * 64];
    #pragma unroll 8
    for (int ic = 0; ic < 64; ++ic) dst[ic] = (_Float16)((float)src[ic] * attn);
}

// ---------------- FC 65536 -> 50 via WMMA (M=batch, N padded to 64) ----------------
// XA row is NHWC-flat; Wfch rows were permuted to match.
__global__ void __launch_bounds__(128) k_fc_wmma(const _Float16* __restrict__ X,
                                                 const _Float16* __restrict__ Wh,
                                                 const float* __restrict__ bfc,
                                                 float* __restrict__ cnnf) {
    const int K = 65536;
    int wave = threadIdx.x >> 5, lane = threadIdx.x & 31;
    int hi = (lane >= 16) ? 1 : 0, lm = lane & 15;
    int m0 = blockIdx.x * 16;
    int row = m0 + lm;
    int oc  = wave * 16 + lm;
    const _Float16* arow = X + (size_t)row * K + hi * 8;
    const _Float16* brow = (oc < 50) ? (Wh + (size_t)oc * K + hi * 8) : nullptr;

    Acc8 acc;
    #pragma unroll
    for (int r = 0; r < 8; ++r) acc.f[r] = 0.f;

    for (int k0 = 0; k0 < K; k0 += 32) {
        Frag16 a, bf;
        const U4* ap = (const U4*)(arow + k0);
        a.q[0] = ap[0];                                    // K = k0 + hi*8 + {0..7}
        a.q[1] = ap[2];                                    // K = k0 + hi*8 + {16..23}
        if (brow) {
            const U4* bp = (const U4*)(brow + k0);
            bf.q[0] = bp[0]; bf.q[1] = bp[2];              // K = k0 + hi*8 + {0..7, 16..23}
        } else {
            #pragma unroll
            for (int e = 0; e < 16; ++e) bf.h[e] = (_Float16)0.f;
        }
        __builtin_prefetch(arow + k0 + 512, 0, 3);         // global_prefetch next slab
        acc.v = __builtin_amdgcn_wmma_f32_16x16x32_f16(false, a.v, false, bf.v,
                                                       (short)0, acc.v, false, false);
    }
    if (oc < 50) {
        float bias = bfc[oc];
        #pragma unroll
        for (int r = 0; r < 8; ++r) {
            int m = m0 + r + hi * 8;
            cnnf[m * 50 + oc] = fmaxf(acc.f[r] + bias, 0.f);
        }
    }
}

// ---------------- handcrafted branch: Sobel edge / brightness / contrast ----------------
__device__ __forceinline__ int refl(int i) { return i < 0 ? -i : (i > 127 ? 254 - i : i); }

__global__ void k_handcrafted(const float* __restrict__ img, float* __restrict__ feat) {
    __shared__ float s0[256], s1[256], s2[256];
    int bc = blockIdx.x;                                   // b*3 + c
    const float* base = img + (size_t)bc * 16384;
    float se = 0.f, ss = 0.f, sq = 0.f;
    for (int p = threadIdx.x; p < 16384; p += blockDim.x) {
        int y = p >> 7, x = p & 127;
        float c0 = floorf(base[p] * 255.f);
        ss += c0; sq += c0 * c0;
        int ym = refl(y - 1), yp = refl(y + 1), xm = refl(x - 1), xp = refl(x + 1);
        float a00 = floorf(base[ym * 128 + xm] * 255.f);
        float a01 = floorf(base[ym * 128 + x ] * 255.f);
        float a02 = floorf(base[ym * 128 + xp] * 255.f);
        float a10 = floorf(base[y  * 128 + xm] * 255.f);
        float a12 = floorf(base[y  * 128 + xp] * 255.f);
        float a20 = floorf(base[yp * 128 + xm] * 255.f);
        float a21 = floorf(base[yp * 128 + x ] * 255.f);
        float a22 = floorf(base[yp * 128 + xp] * 255.f);
        float gx = (a02 + 2.f * a12 + a22) - (a00 + 2.f * a10 + a20);
        float gy = (a20 + 2.f * a21 + a22) - (a00 + 2.f * a01 + a02);
        se += sqrtf(gx * gx + gy * gy);
    }
    s0[threadIdx.x] = se; s1[threadIdx.x] = ss; s2[threadIdx.x] = sq;
    __syncthreads();
    for (int s = 128; s > 0; s >>= 1) {
        if ((int)threadIdx.x < s) {
            s0[threadIdx.x] += s0[threadIdx.x + s];
            s1[threadIdx.x] += s1[threadIdx.x + s];
            s2[threadIdx.x] += s2[threadIdx.x + s];
        }
        __syncthreads();
    }
    if (threadIdx.x == 0) {
        int b = bc / 3, c = bc % 3;
        float inv = 1.f / 16384.f;
        float mean = s1[0] * inv;
        float var  = fmaxf(s2[0] * inv - mean * mean, 0.f);
        feat[b * 9 + c]     = s0[0] * inv;                 // edge
        feat[b * 9 + 3 + c] = mean;                        // brightness
        feat[b * 9 + 6 + c] = sqrtf(var);                  // contrast
    }
}

// ---------------- MLP head: 9->32->64->50, fuse(100->64), classify(64->1), sigmoid ----------
__global__ void k_head(const float* __restrict__ feat, const float* __restrict__ cnnf,
                       const float* Wh1, const float* bh1, const float* Wh2, const float* bh2,
                       const float* Wh3, const float* bh3, const float* Wf,  const float* bf,
                       const float* Wc,  const float* bc,  float* __restrict__ out) {
    __shared__ float sW1[32 * 9],  sb1v[32];
    __shared__ float sW2[64 * 32], sb2v[64];
    __shared__ float sW3[50 * 64], sb3v[50];
    __shared__ float sWf[64 * 100], sbfv[64];
    __shared__ float sWc[64];
    __shared__ float sbc;
    int tid = threadIdx.x;
    for (int i = tid; i < 288;  i += 256) sW1[i] = Wh1[i];
    for (int i = tid; i < 2048; i += 256) sW2[i] = Wh2[i];
    for (int i = tid; i < 3200; i += 256) sW3[i] = Wh3[i];
    for (int i = tid; i < 6400; i += 256) sWf[i] = Wf[i];
    if (tid < 32) sb1v[tid] = bh1[tid];
    if (tid < 64) { sb2v[tid] = bh2[tid]; sbfv[tid] = bf[tid]; sWc[tid] = Wc[tid]; }
    if (tid < 50) sb3v[tid] = bh3[tid];
    if (tid == 0) sbc = bc[0];
    __syncthreads();

    float f[9];
    for (int i = 0; i < 9; ++i) f[i] = feat[tid * 9 + i];
    float cf[50];
    for (int i = 0; i < 50; ++i) cf[i] = cnnf[tid * 50 + i];

    float h1[32];
    for (int j = 0; j < 32; ++j) {
        float a = sb1v[j];
        for (int k = 0; k < 9; ++k) a += f[k] * sW1[j * 9 + k];
        h1[j] = fmaxf(a, 0.f);
    }
    float h2[64];
    for (int j = 0; j < 64; ++j) {
        float a = sb2v[j];
        for (int k = 0; k < 32; ++k) a += h1[k] * sW2[j * 32 + k];
        h2[j] = fmaxf(a, 0.f);
    }
    float h3[50];
    for (int j = 0; j < 50; ++j) {
        float a = sb3v[j];
        for (int k = 0; k < 64; ++k) a += h2[k] * sW3[j * 64 + k];
        h3[j] = fmaxf(a, 0.f);
    }
    float logit = sbc;
    for (int j = 0; j < 64; ++j) {
        float a = sbfv[j];
        for (int k = 0; k < 50; ++k) a += cf[k] * sWf[j * 100 + k];
        for (int k = 0; k < 50; ++k) a += h3[k] * sWf[j * 100 + 50 + k];
        logit += fmaxf(a, 0.f) * sWc[j];
    }
    out[tid] = 1.f / (1.f + expf(-logit));
}

// ---------------- host launcher ----------------
extern "C" void kernel_launch(void* const* d_in, const int* in_sizes, int n_in,
                              void* d_out, int out_size, void* d_ws, size_t ws_size,
                              hipStream_t stream) {
    (void)in_sizes; (void)n_in; (void)out_size; (void)ws_size;
    const float* images = (const float*)d_in[0];
    const float* W1  = (const float*)d_in[1];  const float* b1  = (const float*)d_in[2];
    const float* W2  = (const float*)d_in[3];  const float* b2  = (const float*)d_in[4];
    const float* Wa  = (const float*)d_in[5];  const float* ba  = (const float*)d_in[6];
    const float* Wfc = (const float*)d_in[7];  const float* bfc = (const float*)d_in[8];
    const float* Wh1 = (const float*)d_in[9];  const float* bh1 = (const float*)d_in[10];
    const float* Wh2 = (const float*)d_in[11]; const float* bh2 = (const float*)d_in[12];
    const float* Wh3 = (const float*)d_in[13]; const float* bh3 = (const float*)d_in[14];
    const float* Wf  = (const float*)d_in[15]; const float* bf  = (const float*)d_in[16];
    const float* Wc  = (const float*)d_in[17]; const float* bc  = (const float*)d_in[18];

    char* ws = (char*)d_ws;
    _Float16* P2   = (_Float16*)(ws + 0);                    // NHWC [256,32,32,64] 32 MB
    _Float16* XA   = (_Float16*)(ws + 33554432ull);          // NHWC [256,32,32,64] 32 MB
    _Float16* P1   = (_Float16*)(ws + 67108864ull);          // NHWC [256,64,64,32] 64 MB
    _Float16* W2t  = (_Float16*)(ws + 134217728ull);         // [9,64,32]           36 KB
    _Float16* Wfch = (_Float16*)(ws + 134254592ull);         // [50,65536] perm     6.5 MB
    float*    featv= (float*)   (ws + 140808192ull);         // [256,9]
    float*    cnnf = (float*)   (ws + 140845056ull);         // [256,50]
    float* out = (float*)d_out;

    k_w2_transpose<<<72, 256, 0, stream>>>(W2, W2t);
    k_wfc_perm   <<<12800, 256, 0, stream>>>(Wfc, Wfch);
    k_conv1_pool <<<131072, 256, 0, stream>>>(images, W1, b1, P1);
    k_conv2_wmma <<<NBATCH * 256, 128, 0, stream>>>(P1, W2t, b2, P2);
    k_attn       <<<1024, 256, 0, stream>>>(P2, Wa, ba, XA);
    k_fc_wmma    <<<16, 128, 0, stream>>>(XA, Wfch, bfc, cnnf);
    k_handcrafted<<<768, 256, 0, stream>>>(images, featv);
    k_head       <<<1, 256, 0, stream>>>(featv, cnnf, Wh1, bh1, Wh2, bh2,
                                         Wh3, bh3, Wf, bf, Wc, bc, out);
}
